// MambaMixer_81741817577700
// MI455X (gfx1250) — matrix operand order repaired and checked by reading, not compiled, not run on hardware
//
#include <hip/hip_runtime.h>
#include <hip/hip_bf16.h>
#include <math.h>

// ---------------------------------------------------------------------------
// Mamba mixer for gfx1250 (MI455X).  B=2 L=2048 H=1536 I=3072 N=16 R=96 K=4
//
// GEMMs: v_wmma_f32_16x16x32_bf16, f32 accumulation.  All operands are
// pre-converted to bf16 once (bandwidth-bound elementwise passes), so the
// GEMM inner loop is pure global_load_b128 + v_wmma, software-pipelined,
// with WGP-scope global_prefetch_b8 one k-block ahead.
// Selective scan: one half-wave (16 lanes = N states) per channel,
// cross-lane reduce via __shfl_xor(width=16), t-loop unrolled x4 so loads
// for future timesteps are issued ahead of the dependent exp chain.
// ---------------------------------------------------------------------------

typedef __bf16  v16bf __attribute__((ext_vector_type(16)));
typedef __bf16  v8bf  __attribute__((ext_vector_type(8)));
typedef __bf16  v4bf  __attribute__((ext_vector_type(4)));
typedef float   v8f   __attribute__((ext_vector_type(8)));
typedef float   v4f   __attribute__((ext_vector_type(4)));

static constexpr int Bb = 2, Ll = 2048, Hh = 1536, Ii = 3072;
static constexpr int Nn = 16, Rr = 96, Kc = 4;
static constexpr int ML   = Bb * Ll;        // 4096 rows
static constexpr int TWOI = 2 * Ii;         // 6144

// round-to-nearest-even f32 -> bf16 (bit-twiddle; used only in pre-passes)
__device__ __forceinline__ __bf16 f2bf(float f) {
  union { float f; unsigned u; } v; v.f = f;
  unsigned r = v.u + 0x7FFFu + ((v.u >> 16) & 1u);
  unsigned short h = (unsigned short)(r >> 16);
  return __builtin_bit_cast(__bf16, h);
}

__device__ __forceinline__ v16bf cat16(v8bf lo, v8bf hi) {
  v16bf r;
  #pragma unroll
  for (int j = 0; j < 8; ++j) { r[j] = lo[j]; r[j + 8] = hi[j]; }
  return r;
}

// A-fragment (16x32 bf16): lane l holds row (l&15), K = base+{0..7} and
// base+16+{0..7}, base=(l>>4)*8.  p already includes row*lda + base.
__device__ __forceinline__ v16bf load_a_frag(const __bf16* __restrict__ p) {
  return cat16(*(const v8bf*)(p), *(const v8bf*)(p + 16));
}

// B-fragment (32x16 bf16): lane l holds col (l&15), K = base+{0..7} and
// base+16+{0..7}, base=(l>>4)*8.  p already includes col*ldb + base.
__device__ __forceinline__ v16bf load_b_frag(const __bf16* __restrict__ p) {
  return cat16(*(const v8bf*)(p), *(const v8bf*)(p + 16));
}

#define WMMA_BF16(A_, B_, C_) \
  __builtin_amdgcn_wmma_f32_16x16x32_bf16(false, (A_), false, (B_), (short)0, (C_), false, false)

// C[m,n] = sum_k A[m,k] * W[n,k]   (A, W bf16 row-major; W == B^T)
// One wave owns a 16x64 tile: 4 accumulators share one A fragment.
__global__ __launch_bounds__(128)
void gemm_wmma_bf16(const __bf16* __restrict__ A, int lda,
                    const __bf16* __restrict__ W, int ldb,
                    float* __restrict__ C, int ldc,
                    int M, int K, int tilesN)
{
  int wave = (int)((blockIdx.x * blockDim.x + threadIdx.x) >> 5);
  int lane = (int)(threadIdx.x & 31);
  int totalTiles = (M >> 4) * tilesN;
  if (wave >= totalTiles) return;               // wave-uniform branch

  int tm   = wave / tilesN;
  int tn   = wave % tilesN;
  int m0   = tm << 4;
  int n0   = tn << 6;
  int lrow = lane & 15;
  int half = lane >> 4;

  const __bf16* Abase = A + (size_t)(m0 + lrow) * lda + half * 8;
  const __bf16* B0 = W + (size_t)(n0 + lrow) * ldb + half * 8;
  const __bf16* B1 = B0 + (size_t)16 * ldb;
  const __bf16* B2 = B0 + (size_t)32 * ldb;
  const __bf16* B3 = B0 + (size_t)48 * ldb;

  v8f acc0 = {}, acc1 = {}, acc2 = {}, acc3 = {};

  // software pipeline: current fragments in regs, next loaded before wmma
  v16bf a  = load_a_frag(Abase);
  v16bf b0 = load_b_frag(B0);
  v16bf b1 = load_b_frag(B1);
  v16bf b2 = load_b_frag(B2);
  v16bf b3 = load_b_frag(B3);

  for (int kb = 32; kb < K; kb += 32) {
    // locality=3 -> WGP-scope prefetch (fills near caches on miss)
    __builtin_prefetch(Abase + kb + 32, 0, 3);   // global_prefetch_b8
    __builtin_prefetch(B0 + kb + 32, 0, 3);
    __builtin_prefetch(B2 + kb + 32, 0, 3);
    v16bf an  = load_a_frag(Abase + kb);
    v16bf bn0 = load_b_frag(B0 + kb);
    v16bf bn1 = load_b_frag(B1 + kb);
    v16bf bn2 = load_b_frag(B2 + kb);
    v16bf bn3 = load_b_frag(B3 + kb);
    acc0 = WMMA_BF16(a, b0, acc0);
    acc1 = WMMA_BF16(a, b1, acc1);
    acc2 = WMMA_BF16(a, b2, acc2);
    acc3 = WMMA_BF16(a, b3, acc3);
    a = an; b0 = bn0; b1 = bn1; b2 = bn2; b3 = bn3;
  }
  acc0 = WMMA_BF16(a, b0, acc0);
  acc1 = WMMA_BF16(a, b1, acc1);
  acc2 = WMMA_BF16(a, b2, acc2);
  acc3 = WMMA_BF16(a, b3, acc3);

  // D layout: lanes 0-15 -> M = m0+g, lanes 16-31 -> M = m0+8+g; N = n0+lrow
  float* Crow = C + (size_t)(m0 + half * 8) * ldc + n0 + lrow;
  #pragma unroll
  for (int g = 0; g < 8; ++g) {
    Crow[(size_t)g * ldc + 0]  = acc0[g];
    Crow[(size_t)g * ldc + 16] = acc1[g];
    Crow[(size_t)g * ldc + 32] = acc2[g];
    Crow[(size_t)g * ldc + 48] = acc3[g];
  }
}

// f32 -> bf16 elementwise (4 elems/thread); n must be a multiple of 4
__global__ __launch_bounds__(256)
void cvt_f32_bf16(const float* __restrict__ in, __bf16* __restrict__ out, int n)
{
  int idx = 4 * (int)(blockIdx.x * blockDim.x + threadIdx.x);
  if (idx >= n) return;
  v4f f = *(const v4f*)(in + idx);
  v4bf o;
  o[0] = f2bf(f[0]); o[1] = f2bf(f[1]); o[2] = f2bf(f[2]); o[3] = f2bf(f[3]);
  *(v4bf*)(out + idx) = o;
}

// causal depthwise conv1d (K=4, left pad 3) + SiLU.  x lives in proj[:, 0:I].
// Writes f32 (for the scan's u) and bf16 (GEMM2 operand).
__global__ __launch_bounds__(256)
void conv_silu_kernel(const float* __restrict__ proj,
                      const float* __restrict__ conv_w,
                      const float* __restrict__ conv_b,
                      float* __restrict__ xconv,
                      __bf16* __restrict__ xconv_bf)
{
  int idx = (int)(blockIdx.x * blockDim.x + threadIdx.x);
  if (idx >= ML * Ii) return;
  int i = idx % Ii;
  int m = idx / Ii;
  int l = m % Ll;
  int b = m / Ll;
  const float* w = conv_w + (size_t)i * Kc;
  float acc = conv_b[i];
  #pragma unroll
  for (int j = 0; j < Kc; ++j) {
    int ls = l + j - (Kc - 1);
    if (ls >= 0)
      acc += proj[((size_t)b * Ll + ls) * TWOI + i] * w[j];
  }
  float s = acc / (1.f + __expf(-acc));   // SiLU
  xconv[(size_t)m * Ii + i]    = s;
  xconv_bf[(size_t)m * Ii + i] = f2bf(s);
}

// delta = softplus(dt + dt_bias), in place over proj's x-half (stride 2I)
__global__ __launch_bounds__(256)
void softplus_kernel(float* __restrict__ proj, const float* __restrict__ dt_bias)
{
  int idx = (int)(blockIdx.x * blockDim.x + threadIdx.x);
  if (idx >= ML * Ii) return;
  int i = idx % Ii;
  int m = idx / Ii;
  float* p = proj + (size_t)m * TWOI + i;
  float v = *p + dt_bias[i];
  *p = (v > 20.f) ? v : log1pf(__expf(v));
}

// selective scan: one half-wave (16 lanes = N states) owns one channel i.
// Reads delta (proj x-half), gate (proj gate-half), u (xconv f32), B/C
// (ssm_p f32); writes gated y as bf16 (it is the out_proj GEMM operand).
// t-loop unrolled so per-step loads are issued ahead of the exp chain.
__global__ __launch_bounds__(128)
void scan_kernel(const float* __restrict__ proj,
                 const float* __restrict__ xconv,
                 const float* __restrict__ ssm_p,
                 const float* __restrict__ Amat,   // [I,N]
                 const float* __restrict__ Dvec,   // [I]
                 __bf16* __restrict__ y_bf)        // [ML,I]
{
  int wave = (int)((blockIdx.x * blockDim.x + threadIdx.x) >> 5);
  int lane = (int)(threadIdx.x & 31);
  int half = lane >> 4;
  int n    = lane & 15;
  int b     = wave / (Ii / 2);
  int ipair = wave % (Ii / 2);
  if (b >= Bb) return;
  int i = ipair * 2 + half;

  float An = Amat[(size_t)i * Nn + n];
  float Dv = Dvec[i];
  float s = 0.f;

  #pragma unroll 4
  for (int t = 0; t < Ll; ++t) {
    size_t m = (size_t)b * Ll + t;
    float d  = proj[m * TWOI + i];                 // delta (softplus'ed)
    float u  = xconv[m * Ii + i];
    float Bv = ssm_p[m * 128 + Rr + n];
    float Cv = ssm_p[m * 128 + Rr + Nn + n];
    float g  = proj[m * TWOI + Ii + i];            // gate
    s = s * __expf(d * An) + (d * u) * Bv;
    float y = s * Cv;
    y += __shfl_xor(y, 1, 16);
    y += __shfl_xor(y, 2, 16);
    y += __shfl_xor(y, 4, 16);
    y += __shfl_xor(y, 8, 16);                     // sum over N in all 16 lanes
    float out = (y + u * Dv) * (g / (1.f + __expf(-g)));
    if (n == 0) y_bf[m * Ii + i] = f2bf(out);
  }
}

extern "C" void kernel_launch(void* const* d_in, const int* in_sizes, int n_in,
                              void* d_out, int out_size, void* d_ws, size_t ws_size,
                              hipStream_t stream) {
  (void)in_sizes; (void)n_in; (void)out_size; (void)ws_size;
  const float* hidden  = (const float*)d_in[0];   // [B,L,H]
  const float* W_in    = (const float*)d_in[1];   // [2I,H]
  const float* conv_w  = (const float*)d_in[2];   // [I,K]
  const float* conv_b  = (const float*)d_in[3];   // [I]
  const float* W_x     = (const float*)d_in[4];   // [R+2N, I]
  const float* W_dt    = (const float*)d_in[5];   // [I,R]
  const float* dt_bias = (const float*)d_in[6];   // [I]
  const float* Amat    = (const float*)d_in[7];   // [I,N]
  const float* Dvec    = (const float*)d_in[8];   // [I]
  const float* W_out   = (const float*)d_in[9];   // [H,I]
  float* out = (float*)d_out;                     // [B,L,H]

  // ---- workspace layout -------------------------------------------------
  char* ws = (char*)d_ws;
  size_t cur = 0;
  auto alloc = [&](size_t bytes) { char* p = ws + cur; cur += (bytes + 255) & ~(size_t)255; return p; };
  float*  proj     = (float*) alloc((size_t)ML * TWOI * 4);   // [ML,2I]
  float*  xconv    = (float*) alloc((size_t)ML * Ii * 4);     // [ML,I]
  float*  ssm_p    = (float*) alloc((size_t)ML * 128 * 4);    // [ML,128]
  __bf16* hs_bf    = (__bf16*)alloc((size_t)ML * Hh * 2);
  __bf16* win_bf   = (__bf16*)alloc((size_t)TWOI * Hh * 2);
  __bf16* wx_bf    = (__bf16*)alloc((size_t)128 * Ii * 2);
  __bf16* wdt_bf   = (__bf16*)alloc((size_t)Ii * Rr * 2);
  __bf16* wout_bf  = (__bf16*)alloc((size_t)Hh * Ii * 2);
  __bf16* xconv_bf = (__bf16*)alloc((size_t)ML * Ii * 2);
  __bf16* ssm_bf   = (__bf16*)alloc((size_t)ML * 128 * 2);
  __bf16* y_bf     = (__bf16*)alloc((size_t)ML * Ii * 2);

  const int ew_blocks = (ML * Ii + 255) / 256;
  auto cvt_blocks  = [](int n) { return (n / 4 + 255) / 256; };
  auto gemm_blocks = [](int M, int tilesN) { return ((M >> 4) * tilesN * 32 + 127) / 128; };

  // 0) one-time bf16 conversions of GEMM operands
  cvt_f32_bf16<<<cvt_blocks(ML * Hh),   256, 0, stream>>>(hidden, hs_bf,   ML * Hh);
  cvt_f32_bf16<<<cvt_blocks(TWOI * Hh), 256, 0, stream>>>(W_in,   win_bf,  TWOI * Hh);
  cvt_f32_bf16<<<cvt_blocks(128 * Ii),  256, 0, stream>>>(W_x,    wx_bf,   128 * Ii);
  cvt_f32_bf16<<<cvt_blocks(Ii * Rr),   256, 0, stream>>>(W_dt,   wdt_bf,  Ii * Rr);
  cvt_f32_bf16<<<cvt_blocks(Hh * Ii),   256, 0, stream>>>(W_out,  wout_bf, Hh * Ii);

  // 1) in_proj: proj[ML,2I] = hidden x W_in^T
  gemm_wmma_bf16<<<gemm_blocks(ML, TWOI / 64), 128, 0, stream>>>(
      hs_bf, Hh, win_bf, Hh, proj, TWOI, ML, Hh, TWOI / 64);

  // 2) causal depthwise conv + SiLU -> xconv (f32 + bf16)
  conv_silu_kernel<<<ew_blocks, 256, 0, stream>>>(proj, conv_w, conv_b, xconv, xconv_bf);

  // 3) x_proj: ssm_p[ML,128] = xconv x W_x^T
  gemm_wmma_bf16<<<gemm_blocks(ML, 128 / 64), 128, 0, stream>>>(
      xconv_bf, Ii, wx_bf, Ii, ssm_p, 128, ML, Ii, 128 / 64);

  // 3b) bf16 copy of ssm_p (dt_proj operand)
  cvt_f32_bf16<<<cvt_blocks(ML * 128), 256, 0, stream>>>(ssm_p, ssm_bf, ML * 128);

  // 4) dt_proj: dt[ML,I] = ssm_p[:,0:96] x W_dt^T -> proj x-half (ldc=2I)
  gemm_wmma_bf16<<<gemm_blocks(ML, Ii / 64), 128, 0, stream>>>(
      ssm_bf, 128, wdt_bf, Rr, proj, TWOI, ML, Rr, Ii / 64);

  // 5) delta = softplus(dt + dt_bias), in place
  softplus_kernel<<<ew_blocks, 256, 0, stream>>>(proj, dt_bias);

  // 6) selective scan -> y_bf (gated, D-skip applied)
  scan_kernel<<<(Bb * (Ii / 2) * 32 + 127) / 128, 128, 0, stream>>>(
      proj, xconv, ssm_p, Amat, Dvec, y_bf);

  // 7) out_proj: out[ML,H] = y x W_out^T
  gemm_wmma_bf16<<<gemm_blocks(ML, Hh / 64), 128, 0, stream>>>(
      y_bf, Ii, wout_bf, Ii, out, Hh, ML, ML == ML ? Ii : Ii, Hh / 64);
}
